// GNNEncoder_15496242004285
// MI455X (gfx1250) — matrix operand order repaired and checked by reading, not compiled, not run on hardware
//
#include <hip/hip_runtime.h>
#include <hip/hip_bf16.h>

typedef __attribute__((ext_vector_type(16))) __bf16 v16bf;
typedef __attribute__((ext_vector_type(8)))  float  v8f;

#define EPS 1e-5f
#define DD 512
#define NN 32
#define BB 8

__device__ __forceinline__ __bf16 f2bf(float f) {
    unsigned u = __builtin_bit_cast(unsigned, f);
    unsigned r = u + 0x7FFFu + ((u >> 16) & 1u);   // round-to-nearest-even
    unsigned short h = (unsigned short)(r >> 16);
    return __builtin_bit_cast(__bf16, h);
}

// ---------------------------------------------------------------------------
// Tiled bf16 WMMA GEMM:  C[m,n] = act( sum_k A[m,k] * W[n,k] + bias[n] )
// A: (M,K) row-major f32, W: (N,K) row-major f32 (i.e. C = A * W^T).
// Block: 256 threads (8 wave32), tile 64x64, K-step 64 (4 WMMA / wave / round).
// Row guards are branch-free (clamp + select) to keep the staging loop clean.
// ---------------------------------------------------------------------------
__global__ __launch_bounds__(256) void gemm_nt_wmma(
    const float* __restrict__ A, const float* __restrict__ W,
    const float* __restrict__ bias, float* __restrict__ C,
    int M, int N, int K, int act)
{
    __shared__ __bf16 sA[64 * 64];   // 8 KB
    __shared__ __bf16 sB[64 * 64];   // 8 KB

    const int tid  = threadIdx.x;
    const int lane = tid & 31;
    const int wave = tid >> 5;          // 0..7
    const int wm   = wave >> 1;         // 0..3  (row group of 16)
    const int wn   = wave & 1;          // 0..1  (col group of 32)
    const int half = lane >> 4;         // 0/1
    const int l16  = lane & 15;
    const int mBlk = blockIdx.y * 64;
    const int nBlk = blockIdx.x * 64;

    v8f acc0 = {0.f,0.f,0.f,0.f,0.f,0.f,0.f,0.f};
    v8f acc1 = {0.f,0.f,0.f,0.f,0.f,0.f,0.f,0.f};

    // staging: 64x64 f32 tile = 1024 float4 slots; thread handles 4 slots
    // slot s: row = s >> 4, kc = (s & 15) * 4  -> 16 threads per row, coalesced
    for (int k0 = 0; k0 < K; k0 += 64) {
        #pragma unroll
        for (int u = 0; u < 4; ++u) {
            int s  = tid + u * 256;
            int r  = s >> 4;
            int kc = (s & 15) << 2;
            int gm  = mBlk + r;
            int gmc = gm < M ? gm : M - 1;          // clamp, branch-free
            float4 va = *(const float4*)(A + (size_t)gmc * K + k0 + kc);
            if (gm >= M) va = make_float4(0.f, 0.f, 0.f, 0.f);  // cndmask zero
            sA[r * 64 + kc + 0] = f2bf(va.x);
            sA[r * 64 + kc + 1] = f2bf(va.y);
            sA[r * 64 + kc + 2] = f2bf(va.z);
            sA[r * 64 + kc + 3] = f2bf(va.w);
            int gn  = nBlk + r;
            int gnc = gn < N ? gn : N - 1;
            float4 vb = *(const float4*)(W + (size_t)gnc * K + k0 + kc);
            if (gn >= N) vb = make_float4(0.f, 0.f, 0.f, 0.f);
            sB[r * 64 + kc + 0] = f2bf(vb.x);
            sB[r * 64 + kc + 1] = f2bf(vb.y);
            sB[r * 64 + kc + 2] = f2bf(vb.z);
            sB[r * 64 + kc + 3] = f2bf(vb.w);
        }
        __syncthreads();

        const __bf16* pa  = sA + (wm * 16 + l16) * 64;
        const __bf16* pb0 = sB + (wn * 32 + l16) * 64;
        const __bf16* pb1 = pb0 + 16 * 64;
        #pragma unroll
        for (int ks = 0; ks < 2; ++ks) {
            v16bf af, b0f, b1f;
            // A fragment: 16x32, lane holds row M = l16, K pairs per ISA layout
            #pragma unroll
            for (int v = 0; v < 8; ++v) {
                int kk = ks * 32 + ((v & 4) << 2) + half * 8 + ((v & 3) << 1);
                af[2 * v]     = pa[kk];
                af[2 * v + 1] = pa[kk + 1];
            }
            // B fragments: 32x16, lane half selects K group, l16 selects N col
            #pragma unroll
            for (int v = 0; v < 8; ++v) {
                int kk = ks * 32 + ((v & 4) << 2) + half * 8 + ((v & 3) << 1);
                b0f[2 * v]     = pb0[kk];
                b0f[2 * v + 1] = pb0[kk + 1];
                b1f[2 * v]     = pb1[kk];
                b1f[2 * v + 1] = pb1[kk + 1];
            }
            acc0 = __builtin_amdgcn_wmma_f32_16x16x32_bf16(false, af, false, b0f,
                                                           (short)0, acc0, false, false);
            acc1 = __builtin_amdgcn_wmma_f32_16x16x32_bf16(false, af, false, b1f,
                                                           (short)0, acc1, false, false);
        }
        __syncthreads();
    }

    // C layout: VGPR r -> row half*8+r, col = l16 (+16 for second tile)
    const int n0 = nBlk + wn * 32 + l16;
    const float bi0 = bias ? bias[n0]      : 0.f;
    const float bi1 = bias ? bias[n0 + 16] : 0.f;
    const int mBase = mBlk + wm * 16 + half * 8;
    #pragma unroll
    for (int r = 0; r < 8; ++r) {
        int m = mBase + r;
        if (m >= M) continue;
        float v0 = acc0[r] + bi0;
        float v1 = acc1[r] + bi1;
        if (act) { v0 = fmaxf(v0, 0.f); v1 = fmaxf(v1, 0.f); }
        C[(size_t)m * N + n0]      = v0;
        C[(size_t)m * N + n0 + 16] = v1;
    }
}

// ---------------------------------------------------------------------------
// Elementwise / small helpers
// ---------------------------------------------------------------------------
__global__ void permute_x(const float* __restrict__ x, float* __restrict__ xp) {
    // xp[b, c*NN + n] = x[b, n*DD + c]
    int idx = blockIdx.x * blockDim.x + threadIdx.x;
    if (idx >= BB * NN * DD) return;
    int b = idx / (NN * DD);
    int r = idx % (NN * DD);
    int c = r / NN;
    int n = r % NN;
    xp[idx] = x[(size_t)b * NN * DD + (size_t)n * DD + c];
}

__global__ void vec_add(const float* __restrict__ a, const float* __restrict__ b,
                        float* __restrict__ o, int n) {
    int i = blockIdx.x * blockDim.x + threadIdx.x;
    if (i < n) o[i] = a[i] + b[i];
}

// training-mode BatchNorm over batch dim (8 rows) + ReLU
__global__ void bn_relu(const float* __restrict__ X, const float* __restrict__ g,
                        const float* __restrict__ bta, float* __restrict__ out, int C) {
    int c = blockIdx.x * blockDim.x + threadIdx.x;
    if (c >= C) return;
    float m = 0.f;
    #pragma unroll
    for (int r = 0; r < BB; ++r) m += X[(size_t)r * C + c];
    m *= (1.f / BB);
    float v = 0.f;
    #pragma unroll
    for (int r = 0; r < BB; ++r) { float d = X[(size_t)r * C + c] - m; v += d * d; }
    v *= (1.f / BB);
    float inv = rsqrtf(v + EPS);
    #pragma unroll
    for (int r = 0; r < BB; ++r)
        out[(size_t)r * C + c] = fmaxf((X[(size_t)r * C + c] - m) * inv * g[c] + bta[c], 0.f);
}

__device__ __forceinline__ float blk_sum(float v, float* red) {
    int t = threadIdx.x;
    red[t] = v;
    __syncthreads();
    for (int s = 128; s > 0; s >>= 1) {
        if (t < s) red[t] += red[t + s];
        __syncthreads();
    }
    float r = red[0];
    __syncthreads();
    return r;
}

// LayerNorm over D=512, one block (256 thr) per row.
// mode 0: out = LN(base + y) * g + b
// mode 1: out = relu(base + LN(y) * g + b)
__global__ __launch_bounds__(256) void ln_rows(
    const float* __restrict__ base, const float* __restrict__ y,
    const float* __restrict__ g, const float* __restrict__ bta,
    float* __restrict__ out, int mode)
{
    extern __shared__ float red[];
    size_t row = blockIdx.x;
    int c0 = threadIdx.x * 2;
    float v0, v1;
    if (mode == 0) {
        v0 = base[row * DD + c0]     + y[row * DD + c0];
        v1 = base[row * DD + c0 + 1] + y[row * DD + c0 + 1];
    } else {
        v0 = y[row * DD + c0];
        v1 = y[row * DD + c0 + 1];
    }
    float s  = blk_sum(v0 + v1, red);
    float s2 = blk_sum(v0 * v0 + v1 * v1, red);
    float mean = s * (1.f / DD);
    float var  = s2 * (1.f / DD) - mean * mean;
    float inv  = rsqrtf(var + EPS);
    float o0 = (v0 - mean) * inv * g[c0]     + bta[c0];
    float o1 = (v1 - mean) * inv * g[c0 + 1] + bta[c0 + 1];
    if (mode == 1) {
        o0 = fmaxf(base[row * DD + c0]     + o0, 0.f);
        o1 = fmaxf(base[row * DD + c0 + 1] + o1, 0.f);
    }
    out[row * DD + c0]     = o0;
    out[row * DD + c0 + 1] = o1;
}

// ---------------------------------------------------------------------------
// Fused GNN step (edge LN+ReLU update, sigmoid/softmax-over-j attention,
// channel-wise aggregation, node LN+ReLU). One block per (b,i); the 32x512
// edge slice lives in 64KB LDS (CDNA5: 320KB per WGP).
// ---------------------------------------------------------------------------
__global__ __launch_bounds__(256) void gnn_step_fused(
    const float* __restrict__ Vix, const float* __restrict__ Vjx,
    const float* __restrict__ eproj, int eprojFull,
    const float* __restrict__ eOld, int eOldFull,
    const float* __restrict__ bne_g, const float* __restrict__ bne_b,
    const float* __restrict__ Ujx,  const float* __restrict__ Uxx,
    const float* __restrict__ xbase,
    const float* __restrict__ bnv_g, const float* __restrict__ bnv_b,
    float* __restrict__ e4out, float* __restrict__ xout)
{
    extern __shared__ float smem[];          // 32*512 edge slice + 256 reduce
    float* sE  = smem;
    float* red = smem + NN * DD;

    const int row = blockIdx.x;              // b*NN + i
    const int b   = row >> 5;
    const int c0  = threadIdx.x * 2;

    const float vix0 = Vix[(size_t)row * DD + c0];
    const float vix1 = Vix[(size_t)row * DD + c0 + 1];

    for (int j = 0; j < NN; ++j) {
        size_t ej = ((size_t)row * NN + j) * DD;
        float ep0, ep1;
        if (eprojFull) { ep0 = eproj[ej + c0]; ep1 = eproj[ej + c0 + 1]; }
        else { ep0 = eproj[(size_t)b * DD + c0]; ep1 = eproj[(size_t)b * DD + c0 + 1]; }
        float t0 = vix0 + Vjx[(size_t)(b * NN + j) * DD + c0]     + ep0;
        float t1 = vix1 + Vjx[(size_t)(b * NN + j) * DD + c0 + 1] + ep1;
        float s  = blk_sum(t0 + t1, red);
        float s2 = blk_sum(t0 * t0 + t1 * t1, red);
        float mean = s * (1.f / DD);
        float var  = s2 * (1.f / DD) - mean * mean;
        float inv  = rsqrtf(var + EPS);
        float eo0, eo1;
        if (eOldFull) { eo0 = eOld[ej + c0]; eo1 = eOld[ej + c0 + 1]; }
        else { eo0 = eOld[(size_t)b * DD + c0]; eo1 = eOld[(size_t)b * DD + c0 + 1]; }
        float e0 = eo0 + fmaxf((t0 - mean) * inv * bne_g[c0]     + bne_b[c0],     0.f);
        float e1 = eo1 + fmaxf((t1 - mean) * inv * bne_g[c0 + 1] + bne_b[c0 + 1], 0.f);
        sE[j * DD + c0]     = e0;
        sE[j * DD + c0 + 1] = e1;
        if (e4out) { e4out[ej + c0] = e0; e4out[ej + c0 + 1] = e1; }
    }
    __syncthreads();

    // softmax_j(sigmoid(e)) channel-wise, aggregate Ujx, then node update
    float num0 = 0.f, den0 = 0.f, num1 = 0.f, den1 = 0.f;
    #pragma unroll 4
    for (int j = 0; j < NN; ++j) {
        float sg0 = 1.f / (1.f + __expf(-sE[j * DD + c0]));
        float sg1 = 1.f / (1.f + __expf(-sE[j * DD + c0 + 1]));
        float es0 = __expf(sg0);
        float es1 = __expf(sg1);
        den0 += es0; num0 += es0 * Ujx[(size_t)(b * NN + j) * DD + c0];
        den1 += es1; num1 += es1 * Ujx[(size_t)(b * NN + j) * DD + c0 + 1];
    }
    float xn0 = Uxx[(size_t)row * DD + c0]     + num0 / (den0 * (float)NN);
    float xn1 = Uxx[(size_t)row * DD + c0 + 1] + num1 / (den1 * (float)NN);

    float s  = blk_sum(xn0 + xn1, red);
    float s2 = blk_sum(xn0 * xn0 + xn1 * xn1, red);
    float mean = s * (1.f / DD);
    float var  = s2 * (1.f / DD) - mean * mean;
    float inv  = rsqrtf(var + EPS);
    float o0 = fmaxf(xbase[(size_t)row * DD + c0]     + (xn0 - mean) * inv * bnv_g[c0]     + bnv_b[c0],     0.f);
    float o1 = fmaxf(xbase[(size_t)row * DD + c0 + 1] + (xn1 - mean) * inv * bnv_g[c0 + 1] + bnv_b[c0 + 1], 0.f);
    xout[(size_t)row * DD + c0]     = o0;
    xout[(size_t)row * DD + c0 + 1] = o1;
}

// ---------------------------------------------------------------------------
// Host-side orchestration
// ---------------------------------------------------------------------------
static inline void gemm(const float* A, const float* W, const float* bias, float* C,
                        int M, int N, int K, int act, hipStream_t s) {
    dim3 g(N / 64, (M + 63) / 64);
    gemm_nt_wmma<<<g, 256, 0, s>>>(A, W, bias, C, M, N, K, act);
}

extern "C" void kernel_launch(void* const* d_in, const int* in_sizes, int n_in,
                              void* d_out, int out_size, void* d_ws, size_t ws_size,
                              hipStream_t stream) {
    (void)in_sizes; (void)n_in; (void)out_size; (void)ws_size;
    const int D = DD;
    // input order (setup_inputs dict insertion order):
    const float* X0      = (const float*)d_in[0];
    const float* G_W1    = (const float*)d_in[1];   // (4D, D, N)
    const float* G_B1    = (const float*)d_in[2];
    const float* G_G1    = (const float*)d_in[3];
    const float* G_BE1   = (const float*)d_in[4];
    const float* G_W2    = (const float*)d_in[5];   // (D, 4D)
    const float* G_B2    = (const float*)d_in[6];
    const float* G_G2    = (const float*)d_in[7];
    const float* G_BE2   = (const float*)d_in[8];
    const float* FAM_INW = (const float*)d_in[9];   // (3D, D)
    const float* FAM_INB = (const float*)d_in[10];
    const float* FAM_OW  = (const float*)d_in[11];
    const float* FAM_OB  = (const float*)d_in[12];
    const float* ARM_INW = (const float*)d_in[13];
    const float* ARM_INB = (const float*)d_in[14];
    const float* ARM_OW  = (const float*)d_in[15];
    const float* ARM_OB  = (const float*)d_in[16];
    const float* EP_W1   = (const float*)d_in[17];  // (4D, D)
    const float* EP_B1   = (const float*)d_in[18];
    const float* EP_W2   = (const float*)d_in[19];  // (D, 4D)
    const float* EP_B2   = (const float*)d_in[20];
    const float* EP_LG   = (const float*)d_in[21];
    const float* EP_LB   = (const float*)d_in[22];
    const float* A1 = (const float*)d_in[23];
    const float* B1 = (const float*)d_in[24];
    const float* E1 = (const float*)d_in[25];
    const float* U1 = (const float*)d_in[26];
    const float* V1 = (const float*)d_in[27];
    const float* BNE1G = (const float*)d_in[28];
    const float* BNE1B = (const float*)d_in[29];
    const float* BNV1G = (const float*)d_in[30];
    const float* BNV1B = (const float*)d_in[31];
    const float* A2 = (const float*)d_in[32];
    const float* B2 = (const float*)d_in[33];
    const float* E2 = (const float*)d_in[34];
    const float* U2 = (const float*)d_in[35];
    const float* V2 = (const float*)d_in[36];
    const float* BNE2G = (const float*)d_in[37];
    const float* BNE2B = (const float*)d_in[38];
    const float* BNV2G = (const float*)d_in[39];
    const float* BNV2B = (const float*)d_in[40];
    const float* F_W1 = (const float*)d_in[41];
    const float* F_B1 = (const float*)d_in[42];
    const float* F_W2 = (const float*)d_in[43];
    const float* F_B2 = (const float*)d_in[44];
    const float* F_LG = (const float*)d_in[45];
    const float* F_LB = (const float*)d_in[46];

    float* ws = (float*)d_ws;
    float* XP     = ws + 0;         // 8 x 16384
    float* GF1    = ws + 131072;    // 8 x 2048
    float* GF1R   = ws + 147456;
    float* GF2    = ws + 163840;    // 8 x 512
    float* GF     = ws + 167936;
    float* T1     = ws + 172032;
    float* FEAT1  = ws + 176128;
    float* T2     = ws + 180224;
    float* FEAT2  = ws + 184320;
    float* H8     = ws + 188416;    // 8 x 2048
    float* H8B    = ws + 204800;    // 8 x 512
    float* EVEC   = ws + 208896;
    float* EPROJV = ws + 212992;
    float* VIX    = ws + 217088;    // 256 x 512
    float* VJX    = ws + 348160;
    float* UJX    = ws + 479232;
    float* UXW    = ws + 610304;
    float* XV1    = ws + 741376;
    float* XV2    = ws + 872448;
    float* SBUF   = ws + 1003520;
    float* HF     = ws + 1134592;   // 256 x 2048
    float* H2     = ws + 1658880;   // 256 x 512
    float* E4A    = ws + 1789952;   // 8192 x 512
    float* EPROJ2 = ws + 5984256;   // 8192 x 512
    float* OUT    = (float*)d_out;

    const size_t lnSh  = 256 * sizeof(float);
    const size_t gnnSh = (NN * DD + 256) * sizeof(float);

    // ---- GFEM: full-width conv (as GEMM over permuted x) -> BN -> ReLU -> 1x1 -> BN -> ReLU
    permute_x<<<512, 256, 0, stream>>>(X0, XP);
    gemm(XP, G_W1, G_B1, GF1, BB, 4 * D, NN * D, 0, stream);
    bn_relu<<<8, 256, 0, stream>>>(GF1, G_G1, G_BE1, GF1R, 4 * D);
    gemm(GF1R, G_W2, G_B2, GF2, BB, D, 4 * D, 0, stream);
    bn_relu<<<2, 256, 0, stream>>>(GF2, G_G2, G_BE2, GF, D);

    // ---- GEM (algebraically collapsed): uniform attention -> Wo(Wv gf + bv) + bo
    gemm(GF,    FAM_INW + 2 * D * D, FAM_INB + 2 * D, T1,    BB, D, D, 0, stream);
    gemm(T1,    FAM_OW,              FAM_OB,          FEAT1, BB, D, D, 0, stream);
    gemm(FEAT1, ARM_INW + 2 * D * D, ARM_INB + 2 * D, T2,    BB, D, D, 0, stream);
    gemm(T2,    ARM_OW,              ARM_OB,          FEAT2, BB, D, D, 0, stream);
    // edge FFM on per-batch vector -> evec (broadcast edge tensor)
    gemm(FEAT2, EP_W1, EP_B1, H8,  BB, 4 * D, D, 1, stream);
    gemm(H8,    EP_W2, EP_B2, H8B, BB, D, 4 * D, 0, stream);
    ln_rows<<<BB, 256, lnSh, stream>>>(FEAT2, H8B, EP_LG, EP_LB, EVEC, 0);

    // ---- GNN step 1 (edge tensor still constant over (i,j))
    gemm(X0, A1, nullptr, VIX, BB * NN, D, D, 0, stream);
    gemm(X0, B1, nullptr, VJX, BB * NN, D, D, 0, stream);
    gemm(X0, V1, nullptr, UJX, BB * NN, D, D, 0, stream);
    gemm(X0, U1, nullptr, UXW, BB * NN, D, D, 0, stream);
    gemm(EVEC, E1, nullptr, EPROJV, BB, D, D, 0, stream);
    gnn_step_fused<<<BB * NN, 256, gnnSh, stream>>>(
        VIX, VJX, EPROJV, 0, EVEC, 0, BNE1G, BNE1B,
        UJX, UXW, X0, BNV1G, BNV1B, E4A, XV1);

    // ---- GNN step 2 (full edge tensor)
    gemm(XV1, A2, nullptr, VIX, BB * NN, D, D, 0, stream);
    gemm(XV1, B2, nullptr, VJX, BB * NN, D, D, 0, stream);
    gemm(XV1, V2, nullptr, UJX, BB * NN, D, D, 0, stream);
    gemm(XV1, U2, nullptr, UXW, BB * NN, D, D, 0, stream);
    gemm(E4A, E2, nullptr, EPROJ2, BB * NN * NN, D, D, 0, stream);  // big WMMA GEMM
    gnn_step_fused<<<BB * NN, 256, gnnSh, stream>>>(
        VIX, VJX, EPROJ2, 1, E4A, 1, BNE2G, BNE2B,
        UJX, UXW, XV1, BNV2G, BNV2B, nullptr, XV2);

    // ---- residual + final FFM
    vec_add<<<512, 256, 0, stream>>>(XV2, X0, SBUF, BB * NN * D);
    gemm(SBUF, F_W1, F_B1, HF, BB * NN, 4 * D, D, 1, stream);
    gemm(HF,   F_W2, F_B2, H2, BB * NN, D, 4 * D, 0, stream);
    ln_rows<<<BB * NN, 256, lnSh, stream>>>(SBUF, H2, F_LG, F_LB, OUT, 0);
}
